// FeatureSelectModule_14972255993975
// MI455X (gfx1250) — matrix operand order repaired and checked by reading, not compiled, not run on hardware
//
#include <hip/hip_runtime.h>
#include <hip/hip_bf16.h>
#include <math.h>

typedef __attribute__((ext_vector_type(16))) _Float16     v16h;
typedef __attribute__((ext_vector_type(8)))  float        v8f;
typedef __attribute__((ext_vector_type(8)))  unsigned int v8u;

#define PP   196
#define DD   768
#define NCC  1000
#define KSEL 19

// ---------- helpers ----------
__device__ __forceinline__ float waveSum(float v) {
#pragma unroll
  for (int off = 16; off > 0; off >>= 1) v += __shfl_down(v, off, 32);
  return v;
}

__device__ __forceinline__ float blockSum256(float v, float* s) {
  int tid = threadIdx.x;
  s[tid] = v; __syncthreads();
#pragma unroll
  for (int off = 128; off > 0; off >>= 1) {
    if (tid < off) s[tid] += s[tid + off];
    __syncthreads();
  }
  float r = s[0]; __syncthreads();
  return r;
}

__device__ __forceinline__ float blockMax256(float v, float* s) {
  int tid = threadIdx.x;
  s[tid] = v; __syncthreads();
#pragma unroll
  for (int off = 128; off > 0; off >>= 1) {
    if (tid < off) s[tid] = fmaxf(s[tid], s[tid + off]);
    __syncthreads();
  }
  float r = s[0]; __syncthreads();
  return r;
}

__device__ __forceinline__ float softplusf(float x) {
  return (x > 20.f) ? x : log1pf(expf(x));
}

// ---------- Kernel A: fused dual matvec over 154 MB (bandwidth-bound) ----------
__global__ void fs_matvec_kernel(const float* __restrict__ ori,
                                 const float* __restrict__ w_fs,
                                 const float* __restrict__ w_fs_noise,
                                 const float* __restrict__ noise,
                                 float* __restrict__ t, int N) {
  int lane   = threadIdx.x & 31;
  int wave   = (blockIdx.x * blockDim.x + threadIdx.x) >> 5;
  int nWaves = (gridDim.x * blockDim.x) >> 5;
  const float4* w1 = (const float4*)w_fs;
  const float4* w2 = (const float4*)w_fs_noise;
  for (int row = wave; row < N; row += nWaves) {
    const float4* rp = (const float4*)(ori + (size_t)row * DD);
    float a1 = 0.f, a2 = 0.f;
#pragma unroll
    for (int k = 0; k < 6; ++k) {
      int idx = k * 32 + lane;             // contiguous 512B per k-step (b128 loads)
      float4 o = rp[idx];
      float4 u = w1[idx];
      float4 v = w2[idx];
      a1 += o.x * u.x + o.y * u.y + o.z * u.z + o.w * u.w;
      a2 += o.x * v.x + o.y * v.y + o.z * v.z + o.w * v.w;
    }
    a1 = waveSum(a1);
    a2 = waveSum(a2);
    if (lane == 0) {
      float sdev = softplusf(a2) + 0.01f;
      t[row] = a1 + noise[row] * sdev;
    }
  }
}

// ---------- Kernel B: global softmax stats (single block, data hot in L2) ----------
__global__ void fs_stats_kernel(const float* __restrict__ t, int N,
                                float* __restrict__ stat) {
  __shared__ float s[1024];
  int tid = threadIdx.x;
  float m = -3.4e38f;
  for (int i = tid; i < N; i += 1024) m = fmaxf(m, t[i]);
  s[tid] = m; __syncthreads();
#pragma unroll
  for (int off = 512; off > 0; off >>= 1) {
    if (tid < off) s[tid] = fmaxf(s[tid], s[tid + off]);
    __syncthreads();
  }
  float gmax = s[0]; __syncthreads();
  float sum = 0.f;
  for (int i = tid; i < N; i += 1024) sum += expf(t[i] - gmax);
  s[tid] = sum; __syncthreads();
#pragma unroll
  for (int off = 512; off > 0; off >>= 1) {
    if (tid < off) s[tid] += s[tid + off];
    __syncthreads();
  }
  if (tid == 0) { stat[0] = gmax; stat[1] = s[0]; }
}

// ---------- Kernel C: small tail (top-k, gates, contrast, cos-sim, ent, pool+LN) ----------
__global__ void fs_tail_kernel(const float* __restrict__ ori,
                               const float* __restrict__ aug,
                               const float* __restrict__ ln_gamma,
                               const float* __restrict__ ln_beta,
                               const int* __restrict__ shuffle_idx,
                               const float* __restrict__ t,
                               const float* __restrict__ stat,
                               _Float16* __restrict__ xnh,
                               float* __restrict__ out) {
  __shared__ float sBl[PP];
  __shared__ float sNl[PP];
  __shared__ int   sTop[PP];
  __shared__ int   sSel[KSEL];
  __shared__ float sNorm[KSEL];
  __shared__ float sPooled[DD];
  __shared__ float sRed[256];
  __shared__ float sSim;

  int tid = threadIdx.x;                 // 256 threads = 8 waves
  int wid = tid >> 5, lane = tid & 31;

  float gmax = stat[0], gsum = stat[1];
  if (tid < PP) sBl[tid] = expf(t[tid] - gmax) / gsum;
  if (tid == 0) sSim = 0.f;
  __syncthreads();

  // full descending sort via stable rank (ties: lower index first, matches lax.top_k)
  if (tid < PP) {
    float v = sBl[tid]; int r = 0;
    for (int j = 0; j < PP; ++j) {
      float u = sBl[j];
      if (u > v || (u == v && j < tid)) ++r;
    }
    sTop[r] = tid;
  }
  float sumBl = blockSum256((tid < PP) ? sBl[tid] : 0.f, sRed);  // also fences sTop
  float inv = 1.f / (sumBl + 1e-6f);
  if (tid < PP)  sNl[tid] = sBl[tid] * inv;   // scatter of gates == bl/sum (permutation)
  if (tid < KSEL) sSel[tid] = sTop[shuffle_idx[tid]];
  __syncthreads();

  // contrast = || (aug - ori) * nl ||_F
  float cacc = 0.f;
  for (int i = 0; i < PP; ++i) {
    float w = sNl[i];
    const float* po = ori + (size_t)i * DD;
    const float* pa = aug + (size_t)i * DD;
    for (int d0 = tid; d0 < DD; d0 += 256) {
      float diff = (pa[d0] - po[d0]) * w;
      cacc += diff * diff;
    }
  }
  float ctot = blockSum256(cacc, sRed);
  if (tid == 0) out[NCC + 1] = sqrtf(ctot);

  // norms of the 19 selected rows (wave per row)
  for (int k = wid; k < KSEL; k += 8) {
    const float* rp = ori + (size_t)sSel[k] * DD;
    float acc = 0.f;
    for (int d0 = lane; d0 < DD; d0 += 32) { float x = rp[d0]; acc += x * x; }
    acc = waveSum(acc);
    if (lane == 0) sNorm[k] = fmaxf(sqrtf(acc), 1e-8f);
  }
  __syncthreads();

  // 15 anchor-other dot products: blocks (0,6),(6,12),(12,18)
  for (int pr = wid; pr < 15; pr += 8) {
    int blk = pr / 5;
    int ka = 6 * blk;
    int ko = 6 * blk + 1 + pr % 5;
    const float* pa = ori + (size_t)sSel[ka] * DD;
    const float* po = ori + (size_t)sSel[ko] * DD;
    float acc = 0.f;
    for (int d0 = lane; d0 < DD; d0 += 32) acc += pa[d0] * po[d0];
    acc = waveSum(acc);
    if (lane == 0) atomicAdd(&sSim, acc / (sNorm[ka] * sNorm[ko]));
  }
  __syncthreads();
  if (tid == 0) out[NCC] = sSim / (float)KSEL;

  // entropy term: p = softmax(norm_logits); ent = p0*log(p0)
  float mnl = blockMax256((tid < PP) ? sNl[tid] : -3.4e38f, sRed);
  float Z   = blockSum256((tid < PP) ? expf(sNl[tid] - mnl) : 0.f, sRed);
  if (tid == 0) {
    float p0 = expf(sNl[0] - mnl) / Z;
    out[NCC + 2] = p0 * logf(p0);
  }

  // pooled = sum_i bt[i]*nl[i]; then layernorm -> xn (stored f16 for WMMA)
  for (int d0 = tid; d0 < DD; d0 += 256) {
    float acc = 0.f;
    for (int i = 0; i < PP; ++i) acc += ori[(size_t)i * DD + d0] * sNl[i];
    sPooled[d0] = acc;
  }
  __syncthreads();
  float s1 = 0.f, s2 = 0.f;
  for (int d0 = tid; d0 < DD; d0 += 256) { float x = sPooled[d0]; s1 += x; s2 += x * x; }
  float tot1 = blockSum256(s1, sRed);
  float tot2 = blockSum256(s2, sRed);
  float mu   = tot1 / (float)DD;
  float var  = tot2 / (float)DD - mu * mu;
  float rstd = rsqrtf(var + 1e-6f);
  for (int d0 = tid; d0 < DD; d0 += 256) {
    float xn = (sPooled[d0] - mu) * rstd * ln_gamma[d0] + ln_beta[d0];
    xnh[d0] = (_Float16)xn;
  }
}

// ---------- Kernel D: y = xn @ fc_w + fc_b via v_wmma_f32_16x16x32_f16 ----------
// One wave per 16-column tile. A = xn broadcast into all 16 rows (row 0 used).
// FULL=true: 62 interior tiles, zero branches in the K-loop.
// FULL=false: single tail tile at n0=992 (8 valid cols), compile-time zero-fill.
template <bool FULL>
__global__ void fs_fc_wmma_kernel(const _Float16* __restrict__ xnh,
                                  const float* __restrict__ fc_w,
                                  const float* __restrict__ fc_b,
                                  float* __restrict__ y) {
  int lane  = threadIdx.x;                      // 32 lanes, EXEC all-1s around WMMA
  int n0    = FULL ? (blockIdx.x * 16) : (NCC & ~15);  // tail: 992
  int kHalf = (lane < 16) ? 0 : 8;              // A-fragment K sub-offset (16-bit A 16x32)

  v8f c = {};
#pragma unroll 2
  for (int kk = 0; kk < DD; kk += 32) {
    // A fragment: two runs of 4 dword-aligned half-pairs, k = kHalf+0..7 and 16+kHalf+0..7 (-> 2x b128 loads)
    v8u au;
#pragma unroll
    for (int v = 0; v < 8; ++v) {
      int kbase = (v < 4) ? (kHalf + 2 * v) : (16 + kHalf + 2 * (v - 4));
      au[v] = *(const unsigned int*)(xnh + kk + kbase);
    }
    v16h a = __builtin_bit_cast(v16h, au);

    // B fragment: lane's column N = lane%16, K in same order as A fragment.
    // tail keeps cols 992..999, zero-fills the rest.
    const float* bcol = fc_w + (size_t)kk * NCC + n0 + (lane & 15);
    v16h b;
#pragma unroll
    for (int s = 0; s < 16; ++s) {
      int kb = (s < 8) ? (kHalf + s) : (16 + kHalf + (s - 8));
      b[s] = (FULL || (lane & 15) < (NCC & 15)) ? (_Float16)bcol[(size_t)kb * NCC] : (_Float16)0.f;
    }

    c = __builtin_amdgcn_wmma_f32_16x16x32_f16(
        /*neg_a=*/false, a, /*neg_b=*/false, b,
        /*c_mod=*/(short)0, c, /*reuse_a=*/false, /*reuse_b=*/false);
  }

  // D row M=0 lives in VGPR0 (c[0]) of lanes 0..15, N = lane
  if (lane < (FULL ? 16 : (NCC & 15))) {
    int col = n0 + lane;
    y[col] = c[0] + fc_b[col];
  }
}

extern "C" void kernel_launch(void* const* d_in, const int* in_sizes, int n_in,
                              void* d_out, int out_size, void* d_ws, size_t ws_size,
                              hipStream_t stream) {
  const float* ori        = (const float*)d_in[0];
  const float* aug        = (const float*)d_in[1];
  const float* noise      = (const float*)d_in[2];
  const float* w_fs       = (const float*)d_in[3];
  const float* w_fs_noise = (const float*)d_in[4];
  const float* ln_gamma   = (const float*)d_in[5];
  const float* ln_beta    = (const float*)d_in[6];
  const float* fc_w       = (const float*)d_in[7];
  const float* fc_b       = (const float*)d_in[8];
  const int*   shuf       = (const int*)d_in[9];
  float* out = (float*)d_out;

  int N = in_sizes[0] / DD;   // 50176

  float*    ws_t    = (float*)d_ws;               // N floats
  float*    ws_stat = ws_t + N;                   // 2 floats (padded to 16)
  _Float16* ws_xnh  = (_Float16*)(ws_stat + 16);  // 768 halves

  fs_matvec_kernel<<<512, 256, 0, stream>>>(ori, w_fs, w_fs_noise, noise, ws_t, N);
  fs_stats_kernel<<<1, 1024, 0, stream>>>(ws_t, N, ws_stat);
  fs_tail_kernel<<<1, 256, 0, stream>>>(ori, aug, ln_gamma, ln_beta, shuf,
                                        ws_t, ws_stat, ws_xnh, out);
  fs_fc_wmma_kernel<true ><<<NCC / 16, 32, 0, stream>>>(ws_xnh, fc_w, fc_b, out); // 62 tiles
  fs_fc_wmma_kernel<false><<<1,        32, 0, stream>>>(ws_xnh, fc_w, fc_b, out); // tail tile
}
